// Lfads_11811160064124
// MI455X (gfx1250) — matrix operand order repaired and checked by reading, not compiled, not run on hardware
//
#include <hip/hip_runtime.h>
#include <hip/hip_bf16.h>

typedef __attribute__((ext_vector_type(16))) __bf16 v16bf;
typedef __attribute__((ext_vector_type(8)))  float  v8f;

// ---------------- constants ----------------
static constexpr int T = 256, B = 1024, I = 64, E = 64;

// workspace byte offsets
static constexpr size_t OFF_XBF = 0;                          // bf16 inputs  [T*B*I]
static constexpr size_t OFF_HT  = 33554432;                   // bf16 enc_ht  [T*B*128]
static constexpr size_t OFF_W   = OFF_HT + 67108864;          // bf16 weights (147456 elems)
static constexpr size_t OFF_HO  = OFF_W  + 294912;            // f32 encoder_ho [B*128]
static constexpr size_t OFF_C0  = OFF_HO + 524288;            // f32 controller_ho [B*64]
static constexpr size_t OFF_DH0 = OFF_C0 + 262144;            // f32 decoder_h0 [B*128]
static constexpr size_t OFF_F0  = OFF_DH0 + 524288;           // f32 f0 [B*32]
static constexpr size_t OFF_ACC = OFF_F0 + 131072;            // f32 accums[2]

// d_out element offsets (controls, factors, outputs, vae)
static constexpr size_t FACT_OFF = (size_t)T * B * 3;             // 786432
static constexpr size_t OUT_OFF  = FACT_OFF + (size_t)T * B * 32; // 9175040
static constexpr size_t VAE_OFF  = OUT_OFF + (size_t)T * B * 64;  // 25952256

// bf16 weight block element offsets (within OFF_W)
static constexpr int W_WIHF = 0, W_WHHF = 12288, W_WIHB = 24576, W_WHHB = 36864;
static constexpr int W_CWIH = 49152, W_CWHH = 79872, W_DWHH = 92160;
static constexpr int W_FAC = 141312, W_OUT = 145408;          // end 147456

// ---------------- helpers ----------------
__device__ __forceinline__ unsigned short f2bf(float f) {
  unsigned int u = __float_as_uint(f);
  unsigned int r = (u + 0x7FFFu + ((u >> 16) & 1u)) >> 16;
  return (unsigned short)r;
}
__device__ __forceinline__ float bf2f(unsigned short h) {
  return __uint_as_float(((unsigned int)h) << 16);
}
// fast sigmoid: single v_exp + v_rcp (exact limits: rcp(inf)=0, rcp(1)=1)
__device__ __forceinline__ float sigf(float x) {
  return __builtin_amdgcn_rcpf(1.0f + __expf(-x));
}
// fast tanh: tanh(x) = 1 - 2/(e^{2x}+1); e->inf => 1, e->0 => -1
__device__ __forceinline__ float tanh_fast(float x) {
  return 1.0f - 2.0f * __builtin_amdgcn_rcpf(__expf(2.0f * x) + 1.0f);
}

union Frag { v16bf v; unsigned int u[8]; unsigned short h[16]; };

// A fragment (16x32 bf16, MxK): lane half L<16 -> K 0..7 & 16..23 ; L>=16 -> +8
__device__ __forceinline__ Frag load_frag_A(const unsigned short* src, int ld, int k0, int lane) {
  int r = lane & 15, half = lane >> 4;
  const unsigned short* p = src + r * ld + k0 + half * 8;
  Frag f;
#pragma unroll
  for (int v = 0; v < 4; ++v) {
    f.u[v]     = *(const unsigned int*)(p + 2 * v);
    f.u[v + 4] = *(const unsigned int*)(p + 16 + 2 * v);
  }
  return f;
}
// B fragment (32x16 bf16, KxN) built from W stored row-major [O,K]: B[k,n]=W[col0+n,k]
__device__ __forceinline__ Frag load_frag_B(const unsigned short* W, int ldk, int col0, int k0, int lane) {
  int n = col0 + (lane & 15), half = lane >> 4;
  const unsigned short* p = W + n * ldk + k0 + half * 8;
  Frag f;
#pragma unroll
  for (int v = 0; v < 4; ++v) {
    f.u[v]     = *(const unsigned int*)(p + 2 * v);
    f.u[v + 4] = *(const unsigned int*)(p + 16 + 2 * v);
  }
  return f;
}
__device__ __forceinline__ v8f wmma_bf(const Frag& a, const Frag& b, v8f c) {
  return __builtin_amdgcn_wmma_f32_16x16x32_bf16(false, a.v, false, b.v, (short)0, c, false, false);
}
// C/D layout: lane = n + 16*(m>=8), vgpr v -> m = (lane>=16?8:0)+v
__device__ __forceinline__ void store_acc(float* dst, int ldc, int col0, v8f acc, int lane) {
  int n = col0 + (lane & 15), mb = (lane >> 4) << 3;
#pragma unroll
  for (int v = 0; v < 8; ++v) dst[(mb + v) * ldc + n] = acc[v];
}

// ---------------- prep kernels ----------------
__global__ void prep_inputs(const float* __restrict__ src, unsigned short* __restrict__ dst) {
  size_t n = (size_t)T * B * I;
  for (size_t i = (size_t)blockIdx.x * blockDim.x + threadIdx.x; i < n;
       i += (size_t)gridDim.x * blockDim.x)
    dst[i] = f2bf(src[i]);
}

__global__ void prep_weights(const float* __restrict__ a, const float* __restrict__ b,
                             const float* __restrict__ c, const float* __restrict__ d,
                             const float* __restrict__ e, const float* __restrict__ f,
                             const float* __restrict__ g, const float* __restrict__ h,
                             const float* __restrict__ i2,
                             unsigned short* __restrict__ dst, float* __restrict__ accums) {
  int idx = blockIdx.x * 256 + threadIdx.x;
  if (idx == 0) { accums[0] = 0.f; accums[1] = 0.f; }
  if (idx >= 147456) return;
  int j = idx; float v;
  if (j < 12288) v = a[j];
  else if ((j -= 12288) < 12288) v = b[j];
  else if ((j -= 12288) < 12288) v = c[j];
  else if ((j -= 12288) < 12288) v = d[j];
  else if ((j -= 12288) < 30720) v = e[j];
  else if ((j -= 30720) < 12288) v = f[j];
  else if ((j -= 12288) < 49152) v = g[j];
  else if ((j -= 49152) < 4096)  v = h[j];
  else { j -= 4096; v = i2[j]; }
  dst[idx] = f2bf(v);
}

// ---------------- encoder (fwd+bwd GRU, full time loop in-kernel) ----------------
__global__ __launch_bounds__(256) void enc_kernel(
    const unsigned short* __restrict__ xbf, const unsigned short* __restrict__ wbf,
    const float* __restrict__ bihF, const float* __restrict__ bhhF,
    const float* __restrict__ bihB, const float* __restrict__ bhhB,
    unsigned short* __restrict__ htbf, float* __restrict__ ho) {
  __shared__ __align__(16) unsigned short sWih[192 * 64];
  __shared__ __align__(16) unsigned short sWhh[192 * 64];
  __shared__ __align__(16) unsigned short sH[16 * 64];
  __shared__ float sGi[16 * 192];
  __shared__ float sGh[16 * 192];

  const int tid = threadIdx.x, wid = tid >> 5, lane = tid & 31;
  const int row0 = blockIdx.x * 16;
  const int dir = blockIdx.y;
  const float* bih = dir ? bihB : bihF;
  const float* bhh = dir ? bhhB : bhhF;
  const unsigned short* wih = wbf + (dir ? W_WIHB : W_WIHF);
  const unsigned short* whh = wih + 12288;

  for (int i = tid; i < 6144; i += 256) {
    ((unsigned int*)sWih)[i] = ((const unsigned int*)wih)[i];
    ((unsigned int*)sWhh)[i] = ((const unsigned int*)whh)[i];
  }
  for (int i = tid; i < 512; i += 256) ((unsigned int*)sH)[i] = 0u;
  __syncthreads();

  for (int s = 0; s < T; ++s) {
    const int tt = dir ? (T - 1 - s) : s;
    const unsigned short* xrow = xbf + (size_t)(tt * B + row0) * I;
    // prefetch next step's input rows while this step computes (global_prefetch_b8)
    if (tid < 16 && s + 1 < T) {
      const int ttn = dir ? (T - 2 - s) : (s + 1);
      __builtin_prefetch(xbf + (size_t)(ttn * B + row0 + tid) * I, 0, 1);
    }
    for (int i = wid; i < 24; i += 8) {
      const bool isGi = i < 12;
      const int ct = isGi ? i : i - 12;
      const unsigned short* Asrc = isGi ? xrow : (const unsigned short*)sH;
      const unsigned short* Bsrc = isGi ? (const unsigned short*)sWih : (const unsigned short*)sWhh;
      v8f acc = {};
#pragma unroll
      for (int kk = 0; kk < 64; kk += 32) {
        Frag a = load_frag_A(Asrc, 64, kk, lane);
        Frag b = load_frag_B(Bsrc, 64, ct * 16, kk, lane);
        acc = wmma_bf(a, b, acc);
      }
      store_acc(isGi ? sGi : sGh, 192, ct * 16, acc, lane);
    }
    __syncthreads();
    for (int idx = tid; idx < 1024; idx += 256) {
      int b = idx >> 6, j = idx & 63;
      float ir = sGi[b * 192 + j] + bih[j];
      float iz = sGi[b * 192 + 64 + j] + bih[64 + j];
      float in_ = sGi[b * 192 + 128 + j] + bih[128 + j];
      float hr = sGh[b * 192 + j] + bhh[j];
      float hz = sGh[b * 192 + 64 + j] + bhh[64 + j];
      float hn = sGh[b * 192 + 128 + j] + bhh[128 + j];
      float r = sigf(ir + hr), z = sigf(iz + hz);
      float n = tanh_fast(in_ + r * hn);
      float hv = (1.f - z) * n + z * bf2f(sH[b * 64 + j]);
      sH[b * 64 + j] = f2bf(hv);
      htbf[(size_t)(tt * B + row0 + b) * 128 + dir * 64 + j] = f2bf(hv);
      if (s == T - 1) ho[(size_t)(row0 + b) * 128 + dir * 64 + j] = hv;
    }
    __syncthreads();
  }
}

// ---------------- VAE init ----------------
__global__ __launch_bounds__(256) void init_kernel(
    const float* __restrict__ ho, const float* __restrict__ eps_h0,
    const float* __restrict__ ctrlW, const float* __restrict__ ctrlb,
    const float* __restrict__ mW, const float* __restrict__ mb,
    const float* __restrict__ lvW, const float* __restrict__ lvb,
    const float* __restrict__ facW, const float* __restrict__ facb,
    float* __restrict__ c0g, float* __restrict__ dh0g, float* __restrict__ f0g,
    float* __restrict__ accums) {
  __shared__ float sHo[64 * 128];
  __shared__ float sD[64 * 128];
  const int tid = threadIdx.x, row0 = blockIdx.x * 64;
  for (int i = tid; i < 64 * 128; i += 256) sHo[i] = ho[(size_t)row0 * 128 + i];
  __syncthreads();
  for (int idx = tid; idx < 64 * 64; idx += 256) {
    int b = idx >> 6, j = idx & 63;
    float s = ctrlb[j];
    for (int k = 0; k < 128; ++k) s += sHo[b * 128 + k] * ctrlW[j * 128 + k];
    c0g[(size_t)(row0 + b) * 64 + j] = s;
  }
  float kacc = 0.f;
  for (int idx = tid; idx < 64 * 128; idx += 256) {
    int b = idx >> 7, j = idx & 127;
    float m = mb[j], lv = lvb[j];
    for (int k = 0; k < 128; ++k) {
      float h = sHo[b * 128 + k];
      m += h * mW[j * 128 + k];
      lv += h * lvW[j * 128 + k];
    }
    kacc += -0.5f * (1.f + lv - m * m - __expf(lv));
    float dh = eps_h0[(size_t)(row0 + b) * 128 + j] * __expf(0.5f * lv) + m;
    sD[b * 128 + j] = dh;
    dh0g[(size_t)(row0 + b) * 128 + j] = dh;
  }
  __syncthreads();
  for (int idx = tid; idx < 64 * 32; idx += 256) {
    int b = idx >> 5, j = idx & 31;
    float s = facb[j];
    for (int k = 0; k < 128; ++k) s += sD[b * 128 + k] * facW[j * 128 + k];
    f0g[(size_t)(row0 + b) * 32 + j] = s;
  }
  atomicAdd(&accums[0], kacc);
}

// ---------------- controller + decoder scan (full time loop in-kernel) ----------------
__global__ __launch_bounds__(256) void scan_kernel(
    const unsigned short* __restrict__ htbf, const unsigned short* __restrict__ wbf,
    const float* __restrict__ c0g, const float* __restrict__ dh0g, const float* __restrict__ f0g,
    const float* __restrict__ con_bih, const float* __restrict__ con_bhh,
    const float* __restrict__ cmW, const float* __restrict__ cmb,
    const float* __restrict__ clvW, const float* __restrict__ clvb,
    const float* __restrict__ dWih, const float* __restrict__ dbih, const float* __restrict__ dbhh,
    const float* __restrict__ facb, const float* __restrict__ outb,
    const float* __restrict__ eps_ctrl,
    float* __restrict__ dout, float* __restrict__ accums) {
  __shared__ __align__(16) unsigned short sXcat[16 * 160];
  __shared__ __align__(16) unsigned short sCh[16 * 64];
  __shared__ __align__(16) unsigned short sDh[16 * 128];
  __shared__ __align__(16) unsigned short sF[16 * 32];
  __shared__ float sCgi[16 * 192];
  __shared__ float sCgh[16 * 192];
  __shared__ float sDgh[16 * 384];
  __shared__ float sCtl[16 * 4];

  const int tid = threadIdx.x, wid = tid >> 5, lane = tid & 31;
  const int row0 = blockIdx.x * 16;
  const unsigned short* wCWih = wbf + W_CWIH;
  const unsigned short* wCWhh = wbf + W_CWHH;
  const unsigned short* wDWhh = wbf + W_DWHH;
  const unsigned short* wFac = wbf + W_FAC;
  const unsigned short* wOut = wbf + W_OUT;

  for (int i = tid; i < 16 * 64; i += 256) sCh[i] = f2bf(c0g[(size_t)row0 * 64 + i]);
  for (int i = tid; i < 16 * 128; i += 256) sDh[i] = f2bf(dh0g[(size_t)row0 * 128 + i]);
  for (int i = tid; i < 16 * 32; i += 256) sF[i] = f2bf(f0g[(size_t)row0 * 32 + i]);
  __syncthreads();

  float kacc = 0.f;

  for (int t = 0; t < T; ++t) {
    // 1. xcat = [f | enc_t]   (uint = 2 bf16); prefetch t+1's encoder rows
    if (tid < 32 && t + 1 < T) {
      __builtin_prefetch(
          htbf + (size_t)((t + 1) * B + row0 + (tid >> 1)) * 128 + (tid & 1) * 64, 0, 1);
    }
    for (int i = tid; i < 16 * 80; i += 256) {
      int b = i / 80, c = i % 80;
      unsigned int v;
      if (c < 16) v = ((const unsigned int*)sF)[b * 16 + c];
      else v = *(const unsigned int*)(htbf + (size_t)(t * B + row0 + b) * 128 + (c - 16) * 2);
      ((unsigned int*)sXcat)[b * 80 + c] = v;
    }
    __syncthreads();
    // 2. controller GEMMs
    for (int i = wid; i < 24; i += 8) {
      if (i < 12) {
        v8f acc = {};
#pragma unroll
        for (int kk = 0; kk < 160; kk += 32) {
          Frag a = load_frag_A(sXcat, 160, kk, lane);
          Frag b = load_frag_B(wCWih, 160, i * 16, kk, lane);
          acc = wmma_bf(a, b, acc);
        }
        store_acc(sCgi, 192, i * 16, acc, lane);
      } else {
        int ct = i - 12;
        v8f acc = {};
#pragma unroll
        for (int kk = 0; kk < 64; kk += 32) {
          Frag a = load_frag_A(sCh, 64, kk, lane);
          Frag b = load_frag_B(wCWhh, 64, ct * 16, kk, lane);
          acc = wmma_bf(a, b, acc);
        }
        store_acc(sCgh, 192, ct * 16, acc, lane);
      }
    }
    __syncthreads();
    // 3. controller gate
    for (int idx = tid; idx < 1024; idx += 256) {
      int b = idx >> 6, j = idx & 63;
      float ir = sCgi[b * 192 + j] + con_bih[j];
      float iz = sCgi[b * 192 + 64 + j] + con_bih[64 + j];
      float in_ = sCgi[b * 192 + 128 + j] + con_bih[128 + j];
      float hr = sCgh[b * 192 + j] + con_bhh[j];
      float hz = sCgh[b * 192 + 64 + j] + con_bhh[64 + j];
      float hn = sCgh[b * 192 + 128 + j] + con_bhh[128 + j];
      float r = sigf(ir + hr), z = sigf(iz + hz);
      float n = tanh_fast(in_ + r * hn);
      float cv = (1.f - z) * n + z * bf2f(sCh[b * 64 + j]);
      sCh[b * 64 + j] = f2bf(cv);
    }
    __syncthreads();
    // 4. cm / clv / control + kld
    if (tid < 48) {
      int b = tid / 3, jj = tid % 3;
      float m = cmb[jj], lv = clvb[jj];
      for (int k = 0; k < 64; ++k) {
        float h = bf2f(sCh[b * 64 + k]);
        m += h * cmW[jj * 64 + k];
        lv += h * clvW[jj * 64 + k];
      }
      kacc += -0.5f * (1.f + lv - m * m - __expf(lv));
      float ctl = eps_ctrl[(size_t)(t * B + row0 + b) * 3 + jj] * __expf(0.5f * lv) + m;
      sCtl[b * 4 + jj] = ctl;
      dout[(size_t)(t * B + row0 + b) * 3 + jj] = ctl;
    }
    __syncthreads();
    // 5. decoder gh GEMM (16x384, K=128)
    for (int i = wid; i < 24; i += 8) {
      v8f acc = {};
#pragma unroll
      for (int kk = 0; kk < 128; kk += 32) {
        Frag a = load_frag_A(sDh, 128, kk, lane);
        Frag b = load_frag_B(wDWhh, 128, i * 16, kk, lane);
        acc = wmma_bf(a, b, acc);
      }
      store_acc(sDgh, 384, i * 16, acc, lane);
    }
    __syncthreads();
    // 6. decoder gate (gi via K=3 VALU dots)
    for (int idx = tid; idx < 2048; idx += 256) {
      int b = idx >> 7, j = idx & 127;
      float c0 = sCtl[b * 4 + 0], c1 = sCtl[b * 4 + 1], c2 = sCtl[b * 4 + 2];
      float gir = dbih[j] + c0 * dWih[j * 3] + c1 * dWih[j * 3 + 1] + c2 * dWih[j * 3 + 2];
      float giz = dbih[128 + j] + c0 * dWih[(128 + j) * 3] + c1 * dWih[(128 + j) * 3 + 1] + c2 * dWih[(128 + j) * 3 + 2];
      float gin = dbih[256 + j] + c0 * dWih[(256 + j) * 3] + c1 * dWih[(256 + j) * 3 + 1] + c2 * dWih[(256 + j) * 3 + 2];
      float ghr = sDgh[b * 384 + j] + dbhh[j];
      float ghz = sDgh[b * 384 + 128 + j] + dbhh[128 + j];
      float ghn = sDgh[b * 384 + 256 + j] + dbhh[256 + j];
      float r = sigf(gir + ghr), z = sigf(giz + ghz);
      float n = tanh_fast(gin + r * ghn);
      float dv = (1.f - z) * n + z * bf2f(sDh[b * 128 + j]);
      sDh[b * 128 + j] = f2bf(dv);
    }
    __syncthreads();
    // 7. factors = dh @ fac_W.T + fac_b
    if (wid < 2) {
      v8f acc = {};
#pragma unroll
      for (int kk = 0; kk < 128; kk += 32) {
        Frag a = load_frag_A(sDh, 128, kk, lane);
        Frag b = load_frag_B(wFac, 128, wid * 16, kk, lane);
        acc = wmma_bf(a, b, acc);
      }
      int n = wid * 16 + (lane & 15), mb2 = (lane >> 4) << 3;
#pragma unroll
      for (int v = 0; v < 8; ++v) {
        float fv = acc[v] + facb[n];
        sF[(mb2 + v) * 32 + n] = f2bf(fv);
        dout[FACT_OFF + (size_t)(t * B + row0 + mb2 + v) * 32 + n] = fv;
      }
    }
    __syncthreads();
    // 8. outputs = f @ out_W.T + out_b
    if (wid < 4) {
      v8f acc = {};
      Frag a = load_frag_A(sF, 32, 0, lane);
      Frag b = load_frag_B(wOut, 32, wid * 16, 0, lane);
      acc = wmma_bf(a, b, acc);
      int n = wid * 16 + (lane & 15), mb2 = (lane >> 4) << 3;
#pragma unroll
      for (int v = 0; v < 8; ++v)
        dout[OUT_OFF + (size_t)(t * B + row0 + mb2 + v) * 64 + n] = acc[v] + outb[n];
    }
    __syncthreads();
  }
  if (tid < 48) atomicAdd(&accums[1], kacc);
}

__global__ void fin_kernel(const float* __restrict__ accums, float* __restrict__ dout) {
  dout[VAE_OFF] = accums[0] / (float)B + accums[1] / ((float)B * (float)T);
}

// ---------------- launch ----------------
extern "C" void kernel_launch(void* const* d_in, const int* in_sizes, int n_in,
                              void* d_out, int out_size, void* d_ws, size_t ws_size,
                              hipStream_t stream) {
  (void)in_sizes; (void)n_in; (void)out_size; (void)ws_size;
  char* ws = (char*)d_ws;
  unsigned short* xbf = (unsigned short*)(ws + OFF_XBF);
  unsigned short* htbf = (unsigned short*)(ws + OFF_HT);
  unsigned short* wbf = (unsigned short*)(ws + OFF_W);
  float* ho = (float*)(ws + OFF_HO);
  float* c0g = (float*)(ws + OFF_C0);
  float* dh0g = (float*)(ws + OFF_DH0);
  float* f0g = (float*)(ws + OFF_F0);
  float* accums = (float*)(ws + OFF_ACC);
#define IN(i) ((const float*)d_in[i])

  prep_inputs<<<4096, 256, 0, stream>>>(IN(0), xbf);
  prep_weights<<<576, 256, 0, stream>>>(IN(3), IN(4), IN(7), IN(8), IN(17), IN(18),
                                        IN(26), IN(29), IN(31), wbf, accums);
  enc_kernel<<<dim3(B / 16, 2), 256, 0, stream>>>(xbf, wbf, IN(5), IN(6), IN(9), IN(10),
                                                  htbf, ho);
  init_kernel<<<16, 256, 0, stream>>>(ho, IN(1), IN(11), IN(12), IN(13), IN(14), IN(15),
                                      IN(16), IN(29), IN(30), c0g, dh0g, f0g, accums);
  scan_kernel<<<B / 16, 256, 0, stream>>>(htbf, wbf, c0g, dh0g, f0g, IN(19), IN(20),
                                          IN(21), IN(22), IN(23), IN(24), IN(25), IN(27),
                                          IN(28), IN(30), IN(32), IN(2), (float*)d_out,
                                          accums);
  fin_kernel<<<1, 1, 0, stream>>>(accums, (float*)d_out);
#undef IN
}
